// NodeModel_3375844295136
// MI455X (gfx1250) — compile-verified, matrix operand order checked by reading, not run on hardware
//
#include <hip/hip_runtime.h>
#include <hip/hip_bf16.h>

#define NNODES 50000
#define NEDGES 800000
#define FXDIM  128
#define FEDIM  128
#define HIDDEN 512
#define FOUT   128
#define FIN    256   // FEDIM + FXDIM

typedef __bf16 bf16;
typedef __attribute__((ext_vector_type(16))) __bf16 v16bf;
typedef __attribute__((ext_vector_type(8)))  __bf16 v8bf;
typedef __attribute__((ext_vector_type(8)))  float  v8f;

__device__ __forceinline__ bf16 f2bf(float f) {
  union { float f; unsigned u; } x; x.f = f;
  unsigned u = x.u + 0x7FFFu + ((x.u >> 16) & 1u);   // round-to-nearest-even
  union { unsigned short s; bf16 b; } y;
  y.s = (unsigned short)(u >> 16);
  return y.b;
}

// Load an A-style fragment: 16 bf16 per lane in two contiguous 16-B chunks
// (k = basek+0..7 and basek+16..23). shufflevector concatenation lets the
// backend place both ds_load_b128 results directly into one 8-VGPR fragment.
__device__ __forceinline__ v16bf load_frag_a(const bf16* ap) {
  v8bf lo = *(const v8bf*)(ap);
  v8bf hi = *(const v8bf*)(ap + 16);
  return __builtin_shufflevector(lo, hi, 0, 1, 2, 3, 4, 5, 6, 7,
                                         8, 9, 10, 11, 12, 13, 14, 15);
}

__device__ __forceinline__ v8f wmma_bf16(v16bf a, v16bf b, v8f c) {
  return __builtin_amdgcn_wmma_f32_16x16x32_bf16(false, a, false, b,
                                                 (short)0, c, false, false);
}

__global__ void zero_f32(float* __restrict__ p, int n) {
  int i = blockIdx.x * blockDim.x + threadIdx.x;
  if (i < n) p[i] = 0.0f;
}

// One thread per (edge, group of 4 floats): 4 hardware f32 atomics into L2-resident agg.
__global__ void scatter_edges(const float4* __restrict__ edge_attr,
                              const int* __restrict__ receivers,
                              float* __restrict__ agg) {
  int tid = blockIdx.x * blockDim.x + threadIdx.x;
  if (tid >= NEDGES * 32) return;
  int e = tid >> 5;
  int q = tid & 31;
  float4 v = edge_attr[(size_t)e * 32 + q];
  float* dst = agg + (size_t)receivers[e] * FEDIM + q * 4;
  unsafeAtomicAdd(dst + 0, v.x);
  unsafeAtomicAdd(dst + 1, v.y);
  unsafeAtomicAdd(dst + 2, v.z);
  unsafeAtomicAdd(dst + 3, v.w);
}

// Repack fp32 row-major [K][N] weights into bf16 per-fragment WMMA B layout:
// fragment f = ntile*ktiles + ktile, 512 elems each: [lane(32)][j(16)],
// lane -> n = ntile*16 + lane%16, basek = lane<16 ? 0 : 8,
// j   -> k = ktile*32 + basek + j + (j>=8 ? 8 : 0)   (ISA 7.12.2 16-bit layout)
__global__ void pack_weights(const float* __restrict__ W, bf16* __restrict__ Wp,
                             int K, int N) {
  int ktiles = K >> 5;
  int total = K * N;
  int t = blockIdx.x * blockDim.x + threadIdx.x;
  if (t >= total) return;
  int f = t >> 9;
  int r = t & 511;
  int lane = r >> 4;
  int j = r & 15;
  int ntile = f / ktiles;
  int ktile = f % ktiles;
  int basek = (lane < 16) ? 0 : 8;
  int k = ktile * 32 + basek + j + ((j >= 8) ? 8 : 0);
  int n = ntile * 16 + (lane & 15);
  Wp[t] = f2bf(W[(size_t)k * N + n]);
}

#define XPAD (FIN + 8)      // +16B per row: rotate LDS banks for A-fragment loads
#define HPAD (HIDDEN + 8)

__global__ __launch_bounds__(256) void node_mlp(
    const float* __restrict__ nodes,   // [NNODES, FXDIM]
    const float* __restrict__ agg,     // [NNODES, FEDIM]
    const bf16*  __restrict__ W1p,     // packed fragments, K=256 N=512
    const float* __restrict__ b1,
    const bf16*  __restrict__ W2p,     // packed fragments, K=512 N=128
    const float* __restrict__ b2,
    float* __restrict__ out)           // [NNODES, FOUT] fp32
{
  __shared__ bf16 xTile[32][XPAD];     // ~16.9 KB
  __shared__ bf16 hTile[32][HPAD];     // ~33.3 KB

  const int tid  = threadIdx.x;
  const int lane = tid & 31;
  const int wave = tid >> 5;
  const int nodeBase = blockIdx.x * 32;

  // ---- build x tile = [agg | nodes] in bf16; zero-fill OOB rows ----
  {
    int row = tid >> 3;       // 0..31
    int cg  = tid & 7;        // 8 groups of 32 cols
    int node = nodeBase + row;
    if (node < NNODES) {
      const float* srcA = agg   + (size_t)node * FEDIM;
      const float* srcN = nodes + (size_t)node * FXDIM;
      #pragma unroll
      for (int i = 0; i < 32; ++i) {
        int c = cg * 32 + i;
        float v = (c < FEDIM) ? srcA[c] : srcN[c - FEDIM];
        xTile[row][c] = f2bf(v);
      }
    } else {
      #pragma unroll
      for (int i = 0; i < 32; ++i) xTile[row][cg * 32 + i] = f2bf(0.0f);
    }
  }
  __syncthreads();

  const int basek = (lane < 16) ? 0 : 8;
  const int col16 = lane & 15;
  const int rowHi = (lane < 16) ? 0 : 8;

  // ---- stage A: h = relu(x @ W1 + b1) ----
  // Each wave owns 4 ntiles x 2 mtiles = 8 output tiles; kt is the OUTER loop:
  // per k-step load A0/A1 once (shared by all 4 ntiles), 4 B fragments, and
  // issue 8 independent WMMAs (covers the WMMA->WMMA hazard window for free).
  {
    const bf16* a0rowX = &xTile[col16][0];        // mtile 0, rows 0..15
    const bf16* a1rowX = &xTile[16 + col16][0];   // mtile 1, rows 16..31
    v8f acc[4][2];
    #pragma unroll
    for (int nt = 0; nt < 4; ++nt) {
      float bias = b1[(wave * 4 + nt) * 16 + col16];
      #pragma unroll
      for (int r = 0; r < 8; ++r) { acc[nt][0][r] = bias; acc[nt][1][r] = bias; }
    }
    const bf16* bbase = W1p + (size_t)(wave * 4 * 8) * 512 + lane * 16;
    #pragma unroll
    for (int kt = 0; kt < 8; ++kt) {
      v16bf a0 = load_frag_a(a0rowX + kt * 32 + basek);
      v16bf a1 = load_frag_a(a1rowX + kt * 32 + basek);
      #pragma unroll
      for (int nt = 0; nt < 4; ++nt) {
        v16bf b = *(const v16bf*)(bbase + (size_t)(nt * 8 + kt) * 512);
        acc[nt][0] = wmma_bf16(a0, b, acc[nt][0]);
        acc[nt][1] = wmma_bf16(a1, b, acc[nt][1]);
      }
    }
    #pragma unroll
    for (int nt = 0; nt < 4; ++nt) {
      int ntile = wave * 4 + nt;
      #pragma unroll
      for (int r = 0; r < 8; ++r) {
        float v0 = acc[nt][0][r] > 0.0f ? acc[nt][0][r] : 0.0f;   // fused ReLU
        float v1 = acc[nt][1][r] > 0.0f ? acc[nt][1][r] : 0.0f;
        hTile[rowHi + r][ntile * 16 + col16]      = f2bf(v0);
        hTile[16 + rowHi + r][ntile * 16 + col16] = f2bf(v1);
      }
    }
  }
  __syncthreads();

  // ---- stage B: out = h @ W2 + b2 ----
  // 8 ntiles / 8 waves = 1 ntile per wave; both mtiles share B fragments.
  {
    int ntile = wave;
    float bias = b2[ntile * 16 + col16];
    v8f c0, c1;
    #pragma unroll
    for (int r = 0; r < 8; ++r) { c0[r] = bias; c1[r] = bias; }
    const bf16* a0row = &hTile[col16][0];
    const bf16* a1row = &hTile[16 + col16][0];
    const bf16* bbase = W2p + (size_t)(ntile * 16) * 512 + lane * 16;
    #pragma unroll
    for (int kt = 0; kt < 16; ++kt) {
      v16bf b  = *(const v16bf*)(bbase + (size_t)kt * 512);
      v16bf a0 = load_frag_a(a0row + kt * 32 + basek);
      v16bf a1 = load_frag_a(a1row + kt * 32 + basek);
      c0 = wmma_bf16(a0, b, c0);
      c1 = wmma_bf16(a1, b, c1);
    }
    #pragma unroll
    for (int r = 0; r < 8; ++r) {
      int node0 = nodeBase + rowHi + r;
      int node1 = nodeBase + 16 + rowHi + r;
      if (node0 < NNODES)
        out[(size_t)node0 * FOUT + ntile * 16 + col16] = c0[r];
      if (node1 < NNODES)
        out[(size_t)node1 * FOUT + ntile * 16 + col16] = c1[r];
    }
  }
}

extern "C" void kernel_launch(void* const* d_in, const int* in_sizes, int n_in,
                              void* d_out, int out_size, void* d_ws, size_t ws_size,
                              hipStream_t stream) {
  const float* nodes     = (const float*)d_in[0];
  const float* edge_attr = (const float*)d_in[1];
  // d_in[2] = senders (unused by reference path)
  const int*   receivers = (const int*)d_in[3];
  const float* W1 = (const float*)d_in[4];
  const float* b1 = (const float*)d_in[5];
  const float* W2 = (const float*)d_in[6];
  const float* b2 = (const float*)d_in[7];
  float* out = (float*)d_out;

  char* ws = (char*)d_ws;
  float* agg = (float*)ws;                                   // 50000*128*4 B
  size_t aggBytes = (size_t)NNODES * FEDIM * sizeof(float);
  bf16* W1p = (bf16*)(ws + aggBytes);                        // 256*512*2 B
  bf16* W2p = (bf16*)(ws + aggBytes + (size_t)FIN * HIDDEN * sizeof(bf16));

  int aggN = NNODES * FEDIM;
  zero_f32<<<(aggN + 255) / 256, 256, 0, stream>>>(agg, aggN);

  pack_weights<<<(FIN * HIDDEN + 255) / 256, 256, 0, stream>>>(W1, W1p, FIN, HIDDEN);
  pack_weights<<<(HIDDEN * FOUT + 255) / 256, 256, 0, stream>>>(W2, W2p, HIDDEN, FOUT);

  long total = (long)NEDGES * 32;
  scatter_edges<<<(int)((total + 255) / 256), 256, 0, stream>>>(
      (const float4*)edge_attr, receivers, agg);

  node_mlp<<<(NNODES + 31) / 32, 256, 0, stream>>>(nodes, agg, W1p, b1, W2p, b2, out);
}
